// MLAAttention_53910429499743
// MI455X (gfx1250) — matrix operand order repaired and checked by reading, not compiled, not run on hardware
//
#include <hip/hip_runtime.h>

// ---------------------------------------------------------------------------
// MLA attention forward for MI455X (gfx1250, wave32, WMMA bf16 16x16x32).
// Pipeline: cast -> GEMMs (Qd,Kd,Vd,M^T,Qm,Vup^T) -> flash attn -> out GEMM.
// LDS tiles staged via the CDNA5 Tensor Data Mover, double-buffered so the
// DMA of tile i+1 overlaps the WMMA burst on tile i (TENSORcnt is in-order).
// ---------------------------------------------------------------------------

typedef unsigned short u16;
typedef __attribute__((ext_vector_type(16))) __bf16 v16bf;
typedef __attribute__((ext_vector_type(8)))  float  v8f;

union AFrag { v16bf v; uint4 q[2]; };

__device__ __forceinline__ u16 f2bf(float f) {
  unsigned int u = __float_as_uint(f);
  unsigned int r = u + 0x7fffu + ((u >> 16) & 1u);   // round-to-nearest-even
  return (u16)(r >> 16);
}

__device__ __forceinline__ v8f vzero8() {
  v8f z = {0.f, 0.f, 0.f, 0.f, 0.f, 0.f, 0.f, 0.f};
  return z;
}

// --------------------------- Tensor Data Mover -----------------------------

#if __has_builtin(__builtin_amdgcn_tensor_load_to_lds) && \
    __has_builtin(__builtin_amdgcn_s_wait_tensorcnt)
#define HAVE_TDM 1
typedef __attribute__((ext_vector_type(4))) unsigned int v4u;
typedef __attribute__((ext_vector_type(8))) int v8i;
typedef __attribute__((ext_vector_type(4))) int v4i;

// 2-D tile load: tw elements per row (data_size=2B), th rows, global row
// stride `stride` elements. LDS rows are padded by 16B via the TDM pad unit:
//   padi = 3 -> pad every 64B (tw==32), padi = 4 -> pad every 128B (tw==64).
__device__ __forceinline__ void tdm_load_2d(const u16* g, unsigned ldsAddr,
                                            int stride, int tw, int th,
                                            int padi) {
  unsigned long long ga = (unsigned long long)(size_t)g;
  v4u g0;
  g0[0] = 1u;                                   // count=1, user mode
  g0[1] = ldsAddr;                              // LDS byte address
  g0[2] = (unsigned)(ga & 0xffffffffu);         // global_addr[31:0]
  g0[3] = (unsigned)((ga >> 32) & 0x01ffffffu)  // global_addr[56:32]
          | (2u << 30);                         // type = 2 ("image")
  const unsigned TD = 1u << 22;                 // big logical dims (no OOB)
  v8i g1;
  g1[0] = (int)((1u << 16)        // data_size = 2 bytes
              | (1u << 20)        // pad_enable
              | ((unsigned)padi << 22)
              | (3u << 25));      // pad_amount = 4 dwords (16B)
  g1[1] = (int)((TD & 0xffffu) << 16);                 // tensor_dim0 lo16
  g1[2] = (int)((TD >> 16) | ((TD & 0xffffu) << 16));  // dim0 hi16 | dim1 lo16
  g1[3] = (int)(((TD >> 16) & 0xffffu) | ((unsigned)tw << 16));  // | tile_dim0
  g1[4] = th;                                          // tile_dim1 (tile_dim2=0)
  g1[5] = stride;                                      // tensor_dim0_stride lo32
  g1[6] = 0;                                           // stride hi | dim1_stride
  g1[7] = 0;
  v4i z4 = {0, 0, 0, 0};
#if __clang_major__ >= 23
  v8i z8 = {0, 0, 0, 0, 0, 0, 0, 0};
  __builtin_amdgcn_tensor_load_to_lds(g0, g1, z4, z4, z8, 0);
#else
  __builtin_amdgcn_tensor_load_to_lds(g0, g1, z4, z4, 0);
#endif
}
#else
#define HAVE_TDM 0
#endif

// ------------------------------- cast kernels ------------------------------

__global__ void cast_f32_bf16_kernel(const float* __restrict__ src,
                                     u16* __restrict__ dst, int n) {
  int i = blockIdx.x * blockDim.x + threadIdx.x;
  if (i < n) dst[i] = f2bf(src[i]);
}

// src is [R][C] f32 row-major; dst is [C][R] bf16 (transposed cast).
__global__ void tcast_f32_bf16_kernel(const float* __restrict__ src,
                                      u16* __restrict__ dst, int R, int C) {
  int i = blockIdx.x * blockDim.x + threadIdx.x;
  if (i < R * C) {
    int r = i / C, c = i % C;
    dst[(long long)c * R + r] = f2bf(src[i]);
  }
}

// ------------------------- generic bf16 WMMA GEMM --------------------------
// C[M,N] = scale * A[M,K] @ Bt[N,K]^T (+ bias[N]);  A,Bt bf16 row-major.
// Block = 128x128x32, 256 threads = 8 waves, wave sub-tile 32x64.
// Double-buffered LDS; TDM keeps one k-tile in flight behind the WMMAs.
// All M,N % 128 == 0 and K % 32 == 0 in this pipeline.

__global__ __launch_bounds__(256, 1)
void gemm_bf16_kernel(const u16* __restrict__ A, const u16* __restrict__ Bt,
                      void* __restrict__ Cvoid, const float* __restrict__ bias,
                      int lda, int ldb, int ldc, int K,
                      long long sA, long long sB, long long sC,
                      float scale, int storeBf16) {
  __shared__ u16 As[2][128][40];   // [buf][m][k], +8 pad (TDM pad unit)
  __shared__ u16 Bs[2][128][40];   // [buf][n][k], +8 pad

  const int tid    = threadIdx.x;
  const int lane   = tid & 31;
  const int wave   = tid >> 5;
  const int wm     = wave & 3;          // 4 row-wave groups (32 rows each)
  const int wn     = wave >> 2;         // 2 col-wave groups (64 cols each)
  const int nlan   = lane & 15;
  const int halfUp = lane >> 4;
  const int row0   = blockIdx.y * 128;
  const int col0   = blockIdx.x * 128;

  const u16* Ag = A  + (long long)blockIdx.z * sA + (long long)row0 * lda;
  const u16* Bg = Bt + (long long)blockIdx.z * sB + (long long)col0 * ldb;

#if HAVE_TDM
  const unsigned asAddr[2] = {(unsigned)(size_t)&As[0][0][0],
                              (unsigned)(size_t)&As[1][0][0]};
  const unsigned bsAddr[2] = {(unsigned)(size_t)&Bs[0][0][0],
                              (unsigned)(size_t)&Bs[1][0][0]};
  if (tid < 32) {  // wave 0 drives the TDM: prologue tile 0
    tdm_load_2d(Ag, asAddr[0], lda, 32, 128, 3);
    tdm_load_2d(Bg, bsAddr[0], ldb, 32, 128, 3);
  }
#endif

  v8f acc[2][4];
#pragma unroll
  for (int i = 0; i < 2; ++i)
#pragma unroll
    for (int j = 0; j < 4; ++j) acc[i][j] = vzero8();

  for (int ks = 0; ks < K; ks += 32) {
    const int cur = (ks >> 5) & 1;
#if HAVE_TDM
    if (tid < 32) {
      if (ks + 32 < K) {  // keep next tile in flight, retire current (in-order)
        tdm_load_2d(Ag + ks + 32, asAddr[1 - cur], lda, 32, 128, 3);
        tdm_load_2d(Bg + ks + 32, bsAddr[1 - cur], ldb, 32, 128, 3);
        __builtin_amdgcn_s_wait_tensorcnt(2);
      } else {
        __builtin_amdgcn_s_wait_tensorcnt(0);
      }
    }
#else
#pragma unroll
    for (int i = 0; i < 2; ++i) {
      int vec = tid + i * 256;
      int r   = vec >> 2;
      int c8  = (vec & 3) * 8;
      *(uint4*)&As[cur][r][c8] = *(const uint4*)(Ag + (long long)r * lda + ks + c8);
      *(uint4*)&Bs[cur][r][c8] = *(const uint4*)(Bg + (long long)r * ldb + ks + c8);
    }
    if (ks + 32 < K) {
      __builtin_prefetch(Ag + (long long)(tid >> 1) * lda + ks + 32, 0, 0);
      __builtin_prefetch(Bg + (long long)(tid >> 1) * ldb + ks + 32, 0, 0);
    }
#endif
    __syncthreads();

    AFrag a[2];
#pragma unroll
    for (int mi = 0; mi < 2; ++mi) {
      int m  = wm * 32 + mi * 16 + nlan;
      int kh = halfUp * 8;
      a[mi].q[0] = *(const uint4*)&As[cur][m][kh];
      a[mi].q[1] = *(const uint4*)&As[cur][m][kh + 16];
    }
#pragma unroll
    for (int ni = 0; ni < 4; ++ni) {
      AFrag bfr;
      int n  = wn * 64 + ni * 16 + nlan;
      int k0 = halfUp * 8;
      bfr.q[0] = *(const uint4*)&Bs[cur][n][k0];
      bfr.q[1] = *(const uint4*)&Bs[cur][n][k0 + 16];
#pragma unroll
      for (int mi = 0; mi < 2; ++mi)
        acc[mi][ni] = __builtin_amdgcn_wmma_f32_16x16x32_bf16(
            false, a[mi].v, false, bfr.v, (short)0, acc[mi][ni], false, false);
    }
    __syncthreads();
  }

  float* Cf = (float*)Cvoid + (long long)blockIdx.z * sC;
  u16*   Cb = (u16*)Cvoid   + (long long)blockIdx.z * sC;
#pragma unroll
  for (int mi = 0; mi < 2; ++mi) {
#pragma unroll
    for (int ni = 0; ni < 4; ++ni) {
      int r0 = row0 + wm * 32 + mi * 16 + halfUp * 8;
      int c  = col0 + wn * 64 + ni * 16 + nlan;
      float badd = bias ? bias[c] : 0.f;
#pragma unroll
      for (int v = 0; v < 8; ++v) {
        float val = acc[mi][ni][v] * scale + badd;
        long long idx = (long long)(r0 + v) * ldc + c;
        if (storeBf16) Cb[idx] = f2bf(val);
        else           Cf[idx] = val;
      }
    }
  }
}

// --------------------------- flash attention core --------------------------
// Qm   : [H][B*T][512]  bf16 (already includes 1/sqrt(dh) via M)
// Kd   : [B*T][512]     bf16
// VupT : [H][128][B*T]  bf16 (head-dim major)
// ctx  : [B*T][2048]    bf16 (col = h*128 + dh)
// grid(T/64, H, B), block 128 (4 waves, 16 query rows per wave).

__global__ __launch_bounds__(128, 1)
void mla_flash_kernel(const u16* __restrict__ Qm, const u16* __restrict__ Kd,
                      const u16* __restrict__ VupT, u16* __restrict__ ctx) {
  __shared__ u16 Qs[2][64][40];    // [buf][t][k-chunk]
  __shared__ u16 Ks[2][64][40];    // [buf][s][k-chunk] (== B-matrix layout)
  __shared__ u16 Vs[128][72];      // [dh][s]
  __shared__ u16 Ps[4][16][72];    // per-wave P marshaling (C -> A layout)

  const int tid    = threadIdx.x;
  const int lane   = tid & 31;
  const int wave   = tid >> 5;
  const int nlan   = lane & 15;
  const int halfUp = lane >> 4;
  const int qb = blockIdx.x;
  const int h  = blockIdx.y;
  const int b  = blockIdx.z;
  const int q0 = qb * 64;

  const u16* Qg = Qm + (long long)h * (4096LL * 512) +
                  (long long)(b * 2048 + q0) * 512;
  const u16* Kg = Kd + (long long)(b * 2048) * 512;
  const u16* Vg = VupT + (long long)h * (128LL * 4096) + b * 2048;

#if HAVE_TDM
  const unsigned qsAddr[2] = {(unsigned)(size_t)&Qs[0][0][0],
                              (unsigned)(size_t)&Qs[1][0][0]};
  const unsigned ksAddr[2] = {(unsigned)(size_t)&Ks[0][0][0],
                              (unsigned)(size_t)&Ks[1][0][0]};
  const unsigned vsAddr    = (unsigned)(size_t)&Vs[0][0];
#endif

  v8f   o[8];
  float m_i[8], l_i[8];
#pragma unroll
  for (int i = 0; i < 8; ++i) { o[i] = vzero8(); m_i[i] = -3.0e38f; l_i[i] = 0.f; }

  for (int kb = 0; kb <= qb; ++kb) {          // causal: only s-blocks <= q-block
    const int s0 = kb * 64;
    const u16* Kgb = Kg + (long long)s0 * 512;

    // ---- stage V tile 128(dh) x 64(s) and Q/K tile 0 (TDM in-flight) ----
#if HAVE_TDM
    if (tid < 32) {
      tdm_load_2d(Vg + s0, vsAddr, 4096, 64, 128, 4);
      tdm_load_2d(Qg,  qsAddr[0], 512, 32, 64, 3);
      tdm_load_2d(Kgb, ksAddr[0], 512, 32, 64, 3);
    }
#else
#pragma unroll
    for (int i = 0; i < 8; ++i) {
      int vec = tid + i * 128;
      int dh  = vec >> 3;
      int s8  = (vec & 7) * 8;
      *(uint4*)&Vs[dh][s8] =
          *(const uint4*)(Vg + (long long)dh * 4096 + s0 + s8);
    }
#endif

    // ---- S = Qm @ Kd^T over rk=512 ----
    v8f sa[4];
#pragma unroll
    for (int i = 0; i < 4; ++i) sa[i] = vzero8();

    for (int ks = 0; ks < 512; ks += 32) {
      const int cur = (ks >> 5) & 1;
#if HAVE_TDM
      if (tid < 32) {
        if (ks + 32 < 512) {
          tdm_load_2d(Qg + ks + 32,  qsAddr[1 - cur], 512, 32, 64, 3);
          tdm_load_2d(Kgb + ks + 32, ksAddr[1 - cur], 512, 32, 64, 3);
          __builtin_amdgcn_s_wait_tensorcnt(2);
        } else {
          __builtin_amdgcn_s_wait_tensorcnt(0);
        }
      }
#else
#pragma unroll
      for (int i = 0; i < 2; ++i) {
        int vec = tid + i * 128;
        int r   = vec >> 2;
        int c8  = (vec & 3) * 8;
        *(uint4*)&Qs[cur][r][c8] =
            *(const uint4*)(Qg + (long long)r * 512 + ks + c8);
        *(uint4*)&Ks[cur][r][c8] =
            *(const uint4*)(Kgb + (long long)r * 512 + ks + c8);
      }
#endif
      __syncthreads();

      AFrag a;
      int m  = wave * 16 + nlan;
      int kh = halfUp * 8;
      a.q[0] = *(const uint4*)&Qs[cur][m][kh];
      a.q[1] = *(const uint4*)&Qs[cur][m][kh + 16];
#pragma unroll
      for (int ni = 0; ni < 4; ++ni) {
        AFrag bk;
        int n  = ni * 16 + nlan;
        int k0 = halfUp * 8;
        bk.q[0] = *(const uint4*)&Ks[cur][n][k0];
        bk.q[1] = *(const uint4*)&Ks[cur][n][k0 + 16];
        sa[ni] = __builtin_amdgcn_wmma_f32_16x16x32_bf16(
            false, a.v, false, bk.v, (short)0, sa[ni], false, false);
      }
      __syncthreads();
    }

    // ---- causal mask on diagonal block ----
    if (kb == qb) {
#pragma unroll
      for (int ni = 0; ni < 4; ++ni)
#pragma unroll
        for (int v = 0; v < 8; ++v) {
          int s = s0 + ni * 16 + nlan;
          int q = q0 + wave * 16 + halfUp * 8 + v;
          if (s > q) sa[ni][v] = -3.0e38f;
        }
    }

    // ---- online softmax; row stats shuffle-reduced over 16-lane halves ----
#pragma unroll
    for (int v = 0; v < 8; ++v) {
      float mx = fmaxf(fmaxf(sa[0][v], sa[1][v]), fmaxf(sa[2][v], sa[3][v]));
      mx = fmaxf(mx, __shfl_xor(mx, 1));
      mx = fmaxf(mx, __shfl_xor(mx, 2));
      mx = fmaxf(mx, __shfl_xor(mx, 4));
      mx = fmaxf(mx, __shfl_xor(mx, 8));
      float mn    = fmaxf(m_i[v], mx);
      float alpha = __expf(m_i[v] - mn);
      m_i[v] = mn;
      float rs = 0.f;
#pragma unroll
      for (int ni = 0; ni < 4; ++ni) {
        float p = __expf(sa[ni][v] - mn);
        sa[ni][v] = p;
        rs += p;
      }
      rs += __shfl_xor(rs, 1);
      rs += __shfl_xor(rs, 2);
      rs += __shfl_xor(rs, 4);
      rs += __shfl_xor(rs, 8);
      l_i[v] = l_i[v] * alpha + rs;
#pragma unroll
      for (int ni = 0; ni < 8; ++ni) o[ni][v] *= alpha;
    }

    // ---- re-marshal P: C-layout -> A-fragment layout via LDS ----
#pragma unroll
    for (int ni = 0; ni < 4; ++ni)
#pragma unroll
      for (int v = 0; v < 8; ++v)
        Ps[wave][halfUp * 8 + v][ni * 16 + nlan] = f2bf(sa[ni][v]);
    __syncthreads();

    // ---- O += P (16x64) @ V (64x128) ----
#pragma unroll
    for (int kk = 0; kk < 2; ++kk) {
      AFrag ap;
      int kh = kk * 32 + halfUp * 8;
      ap.q[0] = *(const uint4*)&Ps[wave][nlan][kh];
      ap.q[1] = *(const uint4*)&Ps[wave][nlan][kh + 16];
#pragma unroll
      for (int ni = 0; ni < 8; ++ni) {
        AFrag bv;
        int n  = ni * 16 + nlan;
        int k0 = kk * 32 + halfUp * 8;
        bv.q[0] = *(const uint4*)&Vs[n][k0];
        bv.q[1] = *(const uint4*)&Vs[n][k0 + 16];
        o[ni] = __builtin_amdgcn_wmma_f32_16x16x32_bf16(
            false, ap.v, false, bv.v, (short)0, o[ni], false, false);
      }
    }
    __syncthreads();
  }

  // ---- normalize and emit ctx in [B*T][H*128] bf16 ----
#pragma unroll
  for (int v = 0; v < 8; ++v) {
    float inv = l_i[v] > 0.f ? 1.f / l_i[v] : 0.f;
    int t = b * 2048 + q0 + wave * 16 + halfUp * 8 + v;
#pragma unroll
    for (int ni = 0; ni < 8; ++ni) {
      int col = h * 128 + ni * 16 + nlan;
      ctx[(long long)t * 2048 + col] = f2bf(o[ni][v] * inv);
    }
  }
}

// --------------------------------- launcher --------------------------------

extern "C" void kernel_launch(void* const* d_in, const int* in_sizes, int n_in,
                              void* d_out, int out_size, void* d_ws, size_t ws_size,
                              hipStream_t stream) {
  (void)in_sizes; (void)n_in; (void)out_size; (void)ws_size;

  constexpr int Bb = 2, T = 2048, D = 2048, R = 512, H = 16, DH = 128;
  constexpr int BT = Bb * T;                       // 4096
  const float inv_sqrt_dh = 0.08838834764831845f;  // 1/sqrt(128)

  const float* x   = (const float*)d_in[0];
  const float* qdw = (const float*)d_in[1];
  const float* kdw = (const float*)d_in[2];
  const float* vdw = (const float*)d_in[3];
  const float* quw = (const float*)d_in[4];
  const float* kuw = (const float*)d_in[5];
  const float* vuw = (const float*)d_in[6];
  const float* ow  = (const float*)d_in[7];
  const float* ob  = (const float*)d_in[8];
  float* out = (float*)d_out;

  char* ws = (char*)d_ws;
  size_t off = 0;
  auto take = [&](size_t bytes) -> char* {
    char* p = ws + off;
    off += (bytes + 255) & ~(size_t)255;
    return p;
  };

  u16* xb    = (u16*)take((size_t)BT * D * 2);      // x bf16
  u16* qdb   = (u16*)take((size_t)R * D * 2);       // q_down  [r][d]  (= Bt)
  u16* kdb   = (u16*)take((size_t)R * D * 2);
  u16* vdb   = (u16*)take((size_t)R * D * 2);
  u16* qupT  = (u16*)take((size_t)R * D * 2);       // q_up^T  [r][d]
  u16* kupT  = (u16*)take((size_t)R * D * 2);       // k_up^T  [r][d]
  u16* vub   = (u16*)take((size_t)D * R * 2);       // v_up    [d][r]
  u16* owb   = (u16*)take((size_t)D * D * 2);       // out_w   [n][k]  (= Bt)
  u16* Qd    = (u16*)take((size_t)BT * R * 2);
  u16* Kdn   = (u16*)take((size_t)BT * R * 2);
  u16* Vd    = (u16*)take((size_t)BT * R * 2);
  u16* MhT   = (u16*)take((size_t)H * R * R * 2);   // [h][k][q]
  u16* Qmb   = (u16*)take((size_t)H * BT * R * 2);  // [h][bt][k]
  u16* VupT  = (u16*)take((size_t)H * DH * BT * 2); // [h][dh][bt]
  u16* ctx   = (u16*)take((size_t)BT * D * 2);

  dim3 cblk(256);
  auto cg = [](int n) { return dim3((n + 255) / 256); };

  // ---- casts (only q_up/k_up need a transpose) ----
  cast_f32_bf16_kernel<<<cg(BT * D), cblk, 0, stream>>>(x, xb, BT * D);
  cast_f32_bf16_kernel<<<cg(R * D), cblk, 0, stream>>>(qdw, qdb, R * D);
  cast_f32_bf16_kernel<<<cg(R * D), cblk, 0, stream>>>(kdw, kdb, R * D);
  cast_f32_bf16_kernel<<<cg(R * D), cblk, 0, stream>>>(vdw, vdb, R * D);
  tcast_f32_bf16_kernel<<<cg(D * R), cblk, 0, stream>>>(quw, qupT, D, R);
  tcast_f32_bf16_kernel<<<cg(D * R), cblk, 0, stream>>>(kuw, kupT, D, R);
  cast_f32_bf16_kernel<<<cg(D * R), cblk, 0, stream>>>(vuw, vub, D * R);
  cast_f32_bf16_kernel<<<cg(D * D), cblk, 0, stream>>>(ow, owb, D * D);

  dim3 gblk(256);

  // ---- down projections: [BT,D] @ [R,D]^T -> bf16 ----
  gemm_bf16_kernel<<<dim3(R / 128, BT / 128, 1), gblk, 0, stream>>>(
      xb, qdb, Qd, nullptr, D, D, R, D, 0, 0, 0, 1.f, 1);
  gemm_bf16_kernel<<<dim3(R / 128, BT / 128, 1), gblk, 0, stream>>>(
      xb, kdb, Kdn, nullptr, D, D, R, D, 0, 0, 0, 1.f, 1);
  gemm_bf16_kernel<<<dim3(R / 128, BT / 128, 1), gblk, 0, stream>>>(
      xb, vdb, Vd, nullptr, D, D, R, D, 0, 0, 0, 1.f, 1);

  // ---- M^T_h[k][q] = (Uk_h)^T @ Uq_h * 1/sqrt(dh), per head ----
  gemm_bf16_kernel<<<dim3(R / 128, R / 128, H), gblk, 0, stream>>>(
      kupT, qupT, MhT, nullptr, D, D, R, DH,
      /*sA=*/DH, /*sB=*/DH, /*sC=*/(long long)R * R, inv_sqrt_dh, 1);

  // ---- Qm_h = Qd @ (M^T_h)^T, per head ----
  gemm_bf16_kernel<<<dim3(R / 128, BT / 128, H), gblk, 0, stream>>>(
      Qd, MhT, Qmb, nullptr, R, R, R, R,
      /*sA=*/0, /*sB=*/(long long)R * R, /*sC=*/(long long)BT * R, 1.f, 1);

  // ---- Vup^T_h[dh][bt] = Uv_h @ Vd^T, per head ----
  gemm_bf16_kernel<<<dim3(BT / 128, DH / 128, H), gblk, 0, stream>>>(
      vub, Vd, VupT, nullptr, R, R, BT, R,
      /*sA=*/(long long)DH * R, /*sB=*/0, /*sC=*/(long long)DH * BT, 1.f, 1);

  // ---- fused causal flash attention -> ctx ----
  mla_flash_kernel<<<dim3(T / 64, H, Bb), dim3(128), 0, stream>>>(
      Qmb, Kdn, VupT, ctx);

  // ---- output projection: ctx @ out_w^T + bias -> f32 out ----
  gemm_bf16_kernel<<<dim3(D / 128, BT / 128, 1), gblk, 0, stream>>>(
      ctx, owb, out, ob, D, D, D, D, 0, 0, 0, 1.f, 0);
}
